// GPT_39728447488626
// MI455X (gfx1250) — compile-verified, matrix operand order checked by reading, not run on hardware
//
#include <hip/hip_runtime.h>
#include <hip/hip_bf16.h>
#include <cstdint>

// ---------------------------------------------------------------------------
// GPT forward for MI455X (gfx1250, wave32, WMMA).
// GEMMs: v_wmma_f32_16x16x32_f16 (f16 in / f32 accumulate).
// Activations carry an f16 shadow copy so A-side (and NT B-side) tile fills
// are raw copies -> GLOBAL_LOAD_ASYNC_TO_LDS_B128 (ASYNCcnt) when available.
// Weights stay fp32 and are converted on the LDS-fill path.
// ---------------------------------------------------------------------------

typedef __attribute__((ext_vector_type(16))) _Float16 v16h;
typedef __attribute__((ext_vector_type(8)))  _Float16 v8h;
typedef __attribute__((ext_vector_type(4)))  _Float16 v4h;
typedef __attribute__((ext_vector_type(8)))  float    v8f;
typedef __attribute__((ext_vector_type(4)))  int      v4i;

#define D_   1024
#define T_   2048
#define B_   2
#define L_   4
#define V_   32000
#define DFF_ 4096

constexpr int BM = 128, BN = 128, BK = 32, LDS_K = 40; // +8 halves row pad

#define AS1 __attribute__((address_space(1)))
#define AS3 __attribute__((address_space(3)))

#if defined(__has_builtin)
#if __has_builtin(__builtin_amdgcn_global_load_async_to_lds_b128)
#define HAVE_ASYNC_LDS 1
#endif
#endif
#ifndef HAVE_ASYNC_LDS
#define HAVE_ASYNC_LDS 0
#endif

__device__ __forceinline__ void async_copy_b128(const _Float16* gp, _Float16* lp) {
#if HAVE_ASYNC_LDS
  __builtin_amdgcn_global_load_async_to_lds_b128((AS1 v4i*)gp, (AS3 v4i*)lp,
                                                 0, 0);
#else
  *(v8h*)lp = *(const v8h*)gp;
#endif
}

__device__ __forceinline__ void wait_async0() {
#if HAVE_ASYNC_LDS
#if __has_builtin(__builtin_amdgcn_s_wait_asynccnt)
  __builtin_amdgcn_s_wait_asynccnt(0);
#else
  asm volatile("s_wait_asynccnt 0x0" ::: "memory");
#endif
#endif
}

// ---------------------------------------------------------------------------
// x[b,t,:] = emb[tokens[b,t],:] + pos[t,:]
// ---------------------------------------------------------------------------
__global__ void embed_kernel(const int* __restrict__ tok,
                             const float* __restrict__ emb,
                             const float* __restrict__ pos,
                             float* __restrict__ x) {
  const int row = blockIdx.x;            // b*T + t
  const int t = row & (T_ - 1);
  const int token = tok[row];
  const int d = threadIdx.x * 4;
  const float4 e = *(const float4*)(emb + (size_t)token * D_ + d);
  const float4 p = *(const float4*)(pos + (size_t)t * D_ + d);
  float4 r;
  r.x = e.x + p.x; r.y = e.y + p.y; r.z = e.z + p.z; r.w = e.w + p.w;
  *(float4*)(x + (size_t)row * D_ + d) = r;
}

// ---------------------------------------------------------------------------
// Row-wise LayerNorm over D=1024; writes fp32 master + f16 shadow
// ---------------------------------------------------------------------------
__global__ void layernorm_kernel(const float* __restrict__ x,
                                 const float* __restrict__ g,
                                 const float* __restrict__ b,
                                 float* __restrict__ out32,
                                 _Float16* __restrict__ out16) {
  __shared__ float rs[256];
  __shared__ float rs2[256];
  const int row = blockIdx.x;
  const int tid = threadIdx.x;
  const float* xr = x + (size_t)row * D_;
  float vbuf[4];
  float s = 0.f, s2 = 0.f;
#pragma unroll
  for (int j = 0; j < 4; ++j) {
    const float v = xr[tid + j * 256];
    vbuf[j] = v; s += v; s2 += v * v;
  }
  rs[tid] = s; rs2[tid] = s2;
  __syncthreads();
  for (int st = 128; st > 0; st >>= 1) {
    if (tid < st) { rs[tid] += rs[tid + st]; rs2[tid] += rs2[tid + st]; }
    __syncthreads();
  }
  const float mu   = rs[0] * (1.f / D_);
  const float var  = rs2[0] * (1.f / D_) - mu * mu;
  const float rstd = rsqrtf(var + 1e-5f);
#pragma unroll
  for (int j = 0; j < 4; ++j) {
    const int idx = tid + j * 256;
    const float r = (vbuf[j] - mu) * rstd * g[idx] + b[idx];
    out32[(size_t)row * D_ + idx] = r;
    out16[(size_t)row * D_ + idx] = (_Float16)r;
  }
}

// ---------------------------------------------------------------------------
// Causal softmax on fp32 scores; emits f16 probabilities (zeros for future)
// ---------------------------------------------------------------------------
__global__ void softmax_kernel(const float* __restrict__ att,
                               _Float16* __restrict__ att16) {
  __shared__ float red[256];
  const int row = blockIdx.x;            // b*T + t
  const int tid = threadIdx.x;
  const int t = row & (T_ - 1);
  const float* ar = att + (size_t)row * T_;
  _Float16* aw = att16 + (size_t)row * T_;
  float vals[8];
  float mx = -3.4e38f;
#pragma unroll
  for (int j = 0; j < 8; ++j) {
    const int c = tid + j * 256;
    const float v = (c <= t) ? ar[c] : -3.4e38f;
    vals[j] = v;
    mx = fmaxf(mx, v);
  }
  red[tid] = mx;
  __syncthreads();
  for (int st = 128; st > 0; st >>= 1) {
    if (tid < st) red[tid] = fmaxf(red[tid], red[tid + st]);
    __syncthreads();
  }
  mx = red[0];
  __syncthreads();
  float s = 0.f;
#pragma unroll
  for (int j = 0; j < 8; ++j) {
    const int c = tid + j * 256;
    const float e = (c <= t) ? __expf(vals[j] - mx) : 0.f;
    vals[j] = e; s += e;
  }
  red[tid] = s;
  __syncthreads();
  for (int st = 128; st > 0; st >>= 1) {
    if (tid < st) red[tid] += red[tid + st];
    __syncthreads();
  }
  const float inv = 1.f / red[0];
#pragma unroll
  for (int j = 0; j < 8; ++j) {
    const int c = tid + j * 256;
    aw[c] = (_Float16)(vals[j] * inv);
  }
}

// ---------------------------------------------------------------------------
// WMMA GEMM: C = epilogue(A16[M,K] @ B + bias, resid, scale)
//   BFMT 0: B fp32 [K,N]  (weights, convert-on-fill, prefetched)
//   BFMT 1: B f16  [K,N]  (v)
//   BFMT 2: B f16  [N,K]  (k, A@B^T; async tile copy)
// WF32/WF16: write fp32 C and/or f16 shadow C16.
// 256 threads = 8 wave32 waves (4x2); wave does 2x4 16x16x32 WMMAs per K-tile.
// ---------------------------------------------------------------------------
template <int BFMT, bool BIAS, bool RELU, bool RESID, bool SCALE,
          bool WF32, bool WF16>
__global__ __launch_bounds__(256)
void gemm_wmma_kernel(const _Float16* __restrict__ A, const void* __restrict__ Bp,
                      const float* __restrict__ bias, const float* __restrict__ resid,
                      float* __restrict__ C, _Float16* __restrict__ C16,
                      int M, int N, int K, float scale) {
  __shared__ __align__(16) _Float16 sA[BM * LDS_K];
  __shared__ __align__(16) _Float16 sB[BN * LDS_K]; // stored [n][k]

  const int tid  = threadIdx.x;
  const int lane = tid & 31;
  const int wave = tid >> 5;
  const int wm = wave & 3;   // 4 waves along M (32 rows each)
  const int wn = wave >> 2;  // 2 waves along N (64 cols each)
  const int m0 = blockIdx.y * BM;
  const int n0 = blockIdx.x * BN;

  if constexpr (SCALE) { // causal score GEMM: block fully above diagonal
    if (n0 > m0 + (BM - 1)) return;
  }

  v8f acc[2][4];
#pragma unroll
  for (int i = 0; i < 2; ++i)
#pragma unroll
    for (int j = 0; j < 4; ++j)
#pragma unroll
      for (int r = 0; r < 8; ++r) acc[i][j][r] = 0.f;

  const int ktiles = K / BK;
  const int kbase  = (lane >> 4) * 8;   // ISA 16-bit fragment K-base
  const int fr     = lane & 15;         // fragment row (A) / col (B)

  for (int kt = 0; kt < ktiles; ++kt) {
    // ---- A tile (f16 [M,K]): raw 16B copies -> async DMA to LDS ----
#pragma unroll
    for (int j = 0; j < 2; ++j) {
      const int i  = tid + j * 256;     // 512 x 16B = 8KB tile
      const int r  = i >> 2;            // row in tile
      const int cb = (i & 3) * 8;       // half offset within row
      async_copy_b128(A + (size_t)(m0 + r) * K + kt * BK + cb,
                      sA + r * LDS_K + cb);
    }
    // ---- B tile into sB[n][k] ----
    if constexpr (BFMT == 2) {          // f16 [N,K]: raw copy, async
#pragma unroll
      for (int j = 0; j < 2; ++j) {
        const int i  = tid + j * 256;
        const int r  = i >> 2;          // n index in tile
        const int cb = (i & 3) * 8;
        async_copy_b128((const _Float16*)Bp + (size_t)(n0 + r) * K + kt * BK + cb,
                        sB + r * LDS_K + cb);
      }
    } else if constexpr (BFMT == 1) {   // f16 [K,N]: transpose while filling
      const _Float16* Bh = (const _Float16*)Bp;
      v8h hb[2];
#pragma unroll
      for (int j = 0; j < 2; ++j) {
        const int i  = tid + j * 256;
        const int kk = i >> 4;
        const int nn = (i & 15) * 8;
        hb[j] = *(const v8h*)(Bh + (size_t)(kt * BK + kk) * N + n0 + nn);
      }
#pragma unroll
      for (int j = 0; j < 2; ++j) {
        const int i  = tid + j * 256;
        const int kk = i >> 4;
        const int nn = (i & 15) * 8;
#pragma unroll
        for (int e = 0; e < 8; ++e) sB[(nn + e) * LDS_K + kk] = hb[j][e];
      }
    } else {                            // fp32 [K,N] weights: cvt + transpose
      const float* Bf = (const float*)Bp;
      const int nn = (tid & 31) * 4;
      float4 f[4];
#pragma unroll
      for (int p = 0; p < 4; ++p) {
        const int kk = p * 8 + (tid >> 5);
        f[p] = *(const float4*)(Bf + (size_t)(kt * BK + kk) * N + n0 + nn);
        if (kt + 1 < ktiles)
          __builtin_prefetch(Bf + (size_t)((kt + 1) * BK + kk) * N + n0 + nn, 0, 1);
      }
#pragma unroll
      for (int p = 0; p < 4; ++p) {
        const int kk = p * 8 + (tid >> 5);
        sB[(nn + 0) * LDS_K + kk] = (_Float16)f[p].x;
        sB[(nn + 1) * LDS_K + kk] = (_Float16)f[p].y;
        sB[(nn + 2) * LDS_K + kk] = (_Float16)f[p].z;
        sB[(nn + 3) * LDS_K + kk] = (_Float16)f[p].w;
      }
    }
    wait_async0();        // drain this wave's ASYNCcnt before the barrier
    __syncthreads();

    // ---- fragments (two ds_load_b128 each) + 8 WMMAs ----
    v16h aF[2], bF[4];
#pragma unroll
    for (int mt = 0; mt < 2; ++mt) {
      const _Float16* pA = sA + (wm * 32 + mt * 16 + fr) * LDS_K + kbase;
      const v8h lo = *(const v8h*)pA;
      const v8h hi = *(const v8h*)(pA + 16);
      aF[mt] = __builtin_shufflevector(lo, hi, 0, 1, 2, 3, 4, 5, 6, 7,
                                       8, 9, 10, 11, 12, 13, 14, 15);
    }
#pragma unroll
    for (int nt = 0; nt < 4; ++nt) {
      const _Float16* pB = sB + (wn * 64 + nt * 16 + fr) * LDS_K + kbase;
      const v8h lo = *(const v8h*)pB;
      const v8h hi = *(const v8h*)(pB + 16);
      bF[nt] = __builtin_shufflevector(lo, hi, 0, 1, 2, 3, 4, 5, 6, 7,
                                       8, 9, 10, 11, 12, 13, 14, 15);
    }
#pragma unroll
    for (int mt = 0; mt < 2; ++mt)
#pragma unroll
      for (int nt = 0; nt < 4; ++nt)
        acc[mt][nt] = __builtin_amdgcn_wmma_f32_16x16x32_f16(
            false, aF[mt], false, bF[nt], (short)0, acc[mt][nt], false, false);
    __syncthreads();
  }

  // ---- epilogue: scale, bias, relu, residual; fp32 and/or f16 writes ----
  const int rbase = (lane >> 4) * 8;
  const int cN    = lane & 15;
#pragma unroll
  for (int mt = 0; mt < 2; ++mt) {
#pragma unroll
    for (int nt = 0; nt < 4; ++nt) {
      const int col = n0 + wn * 64 + nt * 16 + cN;
#pragma unroll
      for (int r = 0; r < 8; ++r) {
        const int row = m0 + wm * 32 + mt * 16 + rbase + r;
        float v = acc[mt][nt][r];
        if constexpr (SCALE) v *= scale;
        if constexpr (BIAS)  v += bias[col];
        if constexpr (RELU)  v = fmaxf(v, 0.f);
        if constexpr (RESID) v += resid[(size_t)row * N + col];
        if constexpr (WF32)  C[(size_t)row * N + col] = v;
        if constexpr (WF16)  C16[(size_t)row * N + col] = (_Float16)v;
      }
    }
  }
}

// ---------------------------------------------------------------------------
extern "C" void kernel_launch(void* const* d_in, const int* in_sizes, int n_in,
                              void* d_out, int out_size, void* d_ws, size_t ws_size,
                              hipStream_t stream) {
  (void)in_sizes; (void)n_in; (void)out_size; (void)ws_size;

  const int*   tokens = (const int*)d_in[0];
  const float* emb    = (const float*)d_in[1];
  const float* pos    = (const float*)d_in[2];
  const float* Wq     = (const float*)d_in[3];
  const float* Wk     = (const float*)d_in[4];
  const float* Wv     = (const float*)d_in[5];
  const float* w1     = (const float*)d_in[6];
  const float* b1     = (const float*)d_in[7];
  const float* w2     = (const float*)d_in[8];
  const float* b2     = (const float*)d_in[9];
  const float* g1     = (const float*)d_in[10];
  const float* bln1   = (const float*)d_in[11];
  const float* g2     = (const float*)d_in[12];
  const float* bln2   = (const float*)d_in[13];
  const float* projw  = (const float*)d_in[14];
  const float* projb  = (const float*)d_in[15];
  float* out = (float*)d_out;

  const size_t NTD = (size_t)B_ * T_ * D_;  // 4.19M
  const size_t BTT = (size_t)B_ * T_ * T_;  // 8.39M

  float*    x    = (float*)d_ws;            // fp32 residual stream
  float*    h32  = x + NTD;                 // LN output (fp32 master)
  float*    attf = h32 + NTD;               // fp32 attention scores
  _Float16* x16  = (_Float16*)(attf + BTT); // f16 shadows:
  _Float16* h16  = x16 + NTD;
  _Float16* q16  = h16 + NTD;
  _Float16* k16  = q16 + NTD;
  _Float16* v16  = k16 + NTD;
  _Float16* a16  = v16 + NTD;               // f16 attention probs (B*T*T)
  _Float16* mm16 = q16;                     // FFN mid aliases dead q/k/v/probs

  const int M = B_ * T_;                    // 4096
  const float scl = 0.03125f;               // 1/sqrt(D)

  embed_kernel<<<M, 256, 0, stream>>>(tokens, emb, pos, x);

  for (int l = 0; l < L_; ++l) {
    layernorm_kernel<<<M, 256, 0, stream>>>(x, g1 + (size_t)l * D_,
                                            bln1 + (size_t)l * D_, h32, h16);
    // QKV ([4096,1024] x [1024,1024], fp32 weights) -> f16 shadows only
    gemm_wmma_kernel<0, false, false, false, false, false, true>
        <<<dim3(D_ / BN, M / BM), 256, 0, stream>>>(
            h16, Wq + (size_t)l * D_ * D_, nullptr, nullptr, nullptr, q16,
            M, D_, D_, 1.f);
    gemm_wmma_kernel<0, false, false, false, false, false, true>
        <<<dim3(D_ / BN, M / BM), 256, 0, stream>>>(
            h16, Wk + (size_t)l * D_ * D_, nullptr, nullptr, nullptr, k16,
            M, D_, D_, 1.f);
    gemm_wmma_kernel<0, false, false, false, false, false, true>
        <<<dim3(D_ / BN, M / BM), 256, 0, stream>>>(
            h16, Wv + (size_t)l * D_ * D_, nullptr, nullptr, nullptr, v16,
            M, D_, D_, 1.f);

    // scores = (q @ k^T) * scale  (per batch, NT f16, async fills)
    for (int b = 0; b < B_; ++b)
      gemm_wmma_kernel<2, false, false, false, true, true, false>
          <<<dim3(T_ / BN, T_ / BM), 256, 0, stream>>>(
              q16 + (size_t)b * T_ * D_, k16 + (size_t)b * T_ * D_, nullptr,
              nullptr, attf + (size_t)b * T_ * T_, nullptr, T_, T_, D_, scl);

    softmax_kernel<<<M, 256, 0, stream>>>(attf, a16);

    // x = probs @ v + x  (per batch, NN f16 B, fused residual)
    for (int b = 0; b < B_; ++b)
      gemm_wmma_kernel<1, false, false, true, false, true, false>
          <<<dim3(D_ / BN, T_ / BM), 256, 0, stream>>>(
              a16 + (size_t)b * T_ * T_, v16 + (size_t)b * T_ * D_, nullptr,
              x + (size_t)b * T_ * D_, x + (size_t)b * T_ * D_, nullptr,
              T_, D_, T_, 1.f);

    layernorm_kernel<<<M, 256, 0, stream>>>(x, g2 + (size_t)l * D_,
                                            bln2 + (size_t)l * D_, h32, h16);
    // mm = relu(y @ w1 + b1) -> f16 only
    gemm_wmma_kernel<0, true, true, false, false, false, true>
        <<<dim3(DFF_ / BN, M / BM), 256, 0, stream>>>(
            h16, w1 + (size_t)l * D_ * DFF_, b1 + (size_t)l * DFF_, nullptr,
            nullptr, mm16, M, DFF_, D_, 1.f);
    // x = mm @ w2 + b2 + y -> fp32 master + f16 shadow (shadow feeds proj)
    gemm_wmma_kernel<0, true, false, true, false, true, true>
        <<<dim3(D_ / BN, M / BM), 256, 0, stream>>>(
            mm16, w2 + (size_t)l * DFF_ * D_, b2 + (size_t)l * D_, h32,
            x, x16, M, D_, DFF_, 1.f);
  }

  // logits = x @ proj_w + proj_b  (proj_w = 131MB, resident in 192MB L2)
  gemm_wmma_kernel<0, true, false, false, false, true, false>
      <<<dim3(V_ / BN, M / BM), 256, 0, stream>>>(
          x16, projw, projb, nullptr, out, nullptr, M, V_, D_, 1.f);
}
